// EnhancedMoEModel_24713241821332
// MI455X (gfx1250) — matrix operand order repaired and checked by reading, not compiled, not run on hardware
//
#include <hip/hip_runtime.h>

typedef unsigned short u16;
typedef unsigned int   u32;
typedef __attribute__((ext_vector_type(16))) __bf16 v16bf;
typedef __attribute__((ext_vector_type(8)))  float  v8f;
typedef __attribute__((ext_vector_type(4)))  int    v4i;

#if defined(__gfx1250__) && __has_builtin(__builtin_amdgcn_global_load_async_to_lds_b128)
#define ASYNC_LDS 1
typedef __attribute__((address_space(1))) v4i g_v4i;
typedef __attribute__((address_space(3))) v4i l_v4i;
#endif

// ---------------- helpers ----------------
__device__ __forceinline__ u16 f2bf(float f) {
    u32 u = __float_as_uint(f);
    u = (u + 0x7FFFu + ((u >> 16) & 1u)) >> 16;
    return (u16)u;
}
__device__ __forceinline__ float bf2f(u16 h) {
    return __uint_as_float(((u32)h) << 16);
}

__device__ __forceinline__ void wait_async_lds() {
#ifdef ASYNC_LDS
#if __has_builtin(__builtin_amdgcn_s_wait_asynccnt)
    __builtin_amdgcn_s_wait_asynccnt(0);
#else
    asm volatile("s_wait_asynccnt 0x0" ::: "memory");
#endif
#endif
}

// ---------------- generic batched / segmented bf16 WMMA GEMM ----------------
// C[M,N] (fp32) = A[M,K](bf16) * B (bf16) (+ bias), B either [K,N] (bT=0) or [N,K] (bT=1).
// Block tile 128x64x32, 8 waves, each wave owns a 32x32 C tile (4 WMMAs / K-step).
// Full tiles are staged global->LDS with GLOBAL_LOAD_ASYNC_TO_LDS_B128 (ASYNCcnt),
// edge tiles fall back to guarded VGPR staging.
__global__ __launch_bounds__(256) void gemm_wmma_bf16(
    const u16* __restrict__ Abase, const u16* __restrict__ Bbase,
    float* __restrict__ Cbase, const float* __restrict__ biasBase,
    int M, int N, int K, int lda, int ldb, int ldc, int bT, int zDiv,
    long long sAo, long long sAi, long long sBo, long long sBi,
    long long sCo, long long sCi, long long sBias,
    const int* __restrict__ segCnt, const int* __restrict__ segOff)
{
    __shared__ u16 As[128 * 32];  // [m][k]
    __shared__ u16 Bs[64 * 32];   // [n][k] (transposed tile)

    const int z  = blockIdx.z;
    const int zo = z / zDiv, zi = z - zo * zDiv;
    const u16* A = Abase + zo * sAo + zi * sAi;
    const u16* B = Bbase + zo * sBo + zi * sBi;
    float*     C = Cbase + zo * sCo + zi * sCi;
    const float* bias = biasBase ? (biasBase + (long long)z * sBias) : nullptr;

    int Mz = M;
    if (segCnt) {
        Mz = segCnt[z];
        long long ro = segOff[z];
        A += ro * (long long)lda;
        C += ro * (long long)ldc;
    }

    const int m0 = blockIdx.x * 128;
    const int n0 = blockIdx.y * 64;
    if (m0 >= Mz) return;

    const int tid  = threadIdx.x;
    const int wave = tid >> 5;
    const int lane = tid & 31;
    const int wm = (wave & 3) * 32;     // wave's 32-row slab within 128
    const int wn = (wave >> 2) * 32;    // wave's 32-col slab within 64

    const bool fullA = (m0 + 128 <= Mz);
    const bool fullB = (n0 + 64 <= N);
    (void)fullA; (void)fullB;

    v8f acc00 = {}, acc01 = {}, acc10 = {}, acc11 = {};

    for (int k0 = 0; k0 < K; k0 += 32) {
        // ---- stage A tile: 128 rows x 32 halves (8 KB) ----
#ifdef ASYNC_LDS
        if (fullA) {
#pragma unroll
            for (int i = 0; i < 2; ++i) {
                int idx = tid + i * 256;
                int r = idx >> 2, c = (idx & 3) * 8;
                __builtin_amdgcn_global_load_async_to_lds_b128(
                    (g_v4i*)(A + (long long)(m0 + r) * lda + k0 + c),
                    (l_v4i*)(&As[r * 32 + c]), 0, 0);
            }
        } else
#endif
        {
#pragma unroll
            for (int i = 0; i < 2; ++i) {
                int idx = tid + i * 256;
                int r = idx >> 2, c = (idx & 3) * 8;
                uint4 val = make_uint4(0u, 0u, 0u, 0u);
                int gm = m0 + r;
                if (gm < Mz) val = *(const uint4*)(A + (long long)gm * lda + k0 + c);
                *(uint4*)(&As[r * 32 + c]) = val;
            }
        }
        // ---- stage B tile into [n][k] layout (4 KB) ----
        if (bT) {
            int n = tid >> 2, c = (tid & 3) * 8;
            int gn = n0 + n;
#ifdef ASYNC_LDS
            if (fullB) {
                __builtin_amdgcn_global_load_async_to_lds_b128(
                    (g_v4i*)(B + (long long)gn * ldb + k0 + c),
                    (l_v4i*)(&Bs[n * 32 + c]), 0, 0);
            } else
#endif
            {
                uint4 val = make_uint4(0u, 0u, 0u, 0u);
                if (gn < N) val = *(const uint4*)(B + (long long)gn * ldb + k0 + c);
                *(uint4*)(&Bs[n * 32 + c]) = val;
            }
        } else {
            int k = tid >> 3, c = (tid & 7) * 8;
            int gn = n0 + c;
            union { uint4 q; u16 h[8]; } val;
            val.q = make_uint4(0u, 0u, 0u, 0u);
            if (gn < N) val.q = *(const uint4*)(B + (long long)(k0 + k) * ldb + gn);
#pragma unroll
            for (int j = 0; j < 8; ++j) Bs[(c + j) * 32 + k] = val.h[j];
        }
        if (k0 + 32 < K)
            __builtin_prefetch((const void*)(A + (long long)(m0 + (tid >> 1)) * lda + k0 + 32), 0, 0);
        wait_async_lds();
        __syncthreads();

        // ---- fragments: contiguous k order (k = 16*(lane/16) + s) for both A and B ----
        union frag_t { u32 u[8]; v16bf v; };
        frag_t a0, a1, b0, b1;
        const int am0 = wm + (lane & 15);
        const int am1 = am0 + 16;
        const int kb  = (lane >> 4) * 16;  // A: lanes 0-15 K 0..15; lanes 16-31 K 16..31
#pragma unroll
        for (int i = 0; i < 8; ++i) {
            int kp = kb + 2 * i;
            a0.u[i] = *(const u32*)(&As[am0 * 32 + kp]);
            a1.u[i] = *(const u32*)(&As[am1 * 32 + kp]);
        }
        const int bn = lane & 15;
        const int kk = (lane >> 4) * 16;  // B: lanes 0-15 K 0..15; lanes 16-31 K 16..31
#pragma unroll
        for (int i = 0; i < 8; ++i) {
            b0.u[i] = *(const u32*)(&Bs[(wn + bn) * 32 + kk + 2 * i]);
            b1.u[i] = *(const u32*)(&Bs[(wn + 16 + bn) * 32 + kk + 2 * i]);
        }
        acc00 = __builtin_amdgcn_wmma_f32_16x16x32_bf16(false, a0.v, false, b0.v, (short)0, acc00, false, false);
        acc01 = __builtin_amdgcn_wmma_f32_16x16x32_bf16(false, a0.v, false, b1.v, (short)0, acc01, false, false);
        acc10 = __builtin_amdgcn_wmma_f32_16x16x32_bf16(false, a1.v, false, b0.v, (short)0, acc10, false, false);
        acc11 = __builtin_amdgcn_wmma_f32_16x16x32_bf16(false, a1.v, false, b1.v, (short)0, acc11, false, false);
        __syncthreads();
    }

    // ---- store C per 16x16 f32 D layout: VGPR r -> M = r + 8*(lane>=16), N = lane%16 ----
    const int cn0 = n0 + wn + (lane & 15);
    const int cn1 = cn0 + 16;
    const int mr0 = m0 + wm + (lane >> 4) * 8;
    const int mr1 = mr0 + 16;
    const float bv0 = (bias && cn0 < N) ? bias[cn0] : 0.f;
    const float bv1 = (bias && cn1 < N) ? bias[cn1] : 0.f;
#pragma unroll
    for (int r = 0; r < 8; ++r) {
        int g0 = mr0 + r, g1 = mr1 + r;
        if (g0 < Mz) {
            if (cn0 < N) C[(long long)g0 * ldc + cn0] = acc00[r] + bv0;
            if (cn1 < N) C[(long long)g0 * ldc + cn1] = acc01[r] + bv1;
        }
        if (g1 < Mz) {
            if (cn0 < N) C[(long long)g1 * ldc + cn0] = acc10[r] + bv0;
            if (cn1 < N) C[(long long)g1 * ldc + cn1] = acc11[r] + bv1;
        }
    }
}

// ---------------- elementwise / small kernels ----------------
__global__ void cvt_f32_bf16(const float* __restrict__ in, u16* __restrict__ out, long long n) {
    long long i = (long long)blockIdx.x * blockDim.x + threadIdx.x;
    if (i < n) out[i] = f2bf(in[i]);
}

__global__ void embed_kernel(const int* __restrict__ ids, const float* __restrict__ emb,
                             float* __restrict__ x, int H) {
    int row = blockIdx.x;
    long long base = (long long)ids[row] * H;
    for (int c = threadIdx.x; c < H; c += blockDim.x)
        x[(long long)row * H + c] = emb[base + c];
}

__global__ void ln_kernel(const float* __restrict__ x, const float* __restrict__ w,
                          const float* __restrict__ b, u16* __restrict__ out, int H) {
    int row = blockIdx.x, lane = threadIdx.x;
    const float* xr = x + (long long)row * H;
    float s = 0.f;
    for (int j = lane; j < H; j += 32) s += xr[j];
#pragma unroll
    for (int t = 16; t > 0; t >>= 1) s += __shfl_xor(s, t);
    float m = s / H;
    float v = 0.f;
    for (int j = lane; j < H; j += 32) { float d = xr[j] - m; v += d * d; }
#pragma unroll
    for (int t = 16; t > 0; t >>= 1) v += __shfl_xor(v, t);
    float inv = rsqrtf(v / H + 1e-5f);
    for (int j = lane; j < H; j += 32)
        out[(long long)row * H + j] = f2bf((xr[j] - m) * inv * w[j] + b[j]);
}

__global__ void rope_kernel(const float* __restrict__ q, u16* __restrict__ out,
                            int S, int H, long long total) {
    long long i = (long long)blockIdx.x * blockDim.x + threadIdx.x;
    if (i >= total) return;
    int c = (int)(i % H);
    int n = (int)(i / H);
    int s = n % S;
    int d = c & 63;
    int dh = d & 31;
    float inv = __powf(10000.f, -(float)(2 * dh) / 64.f);
    float ang = (float)s * inv;
    float sn, cs;
    __sincosf(ang, &sn, &cs);
    float v0 = q[i];
    float other = (d < 32) ? -q[i + 32] : q[i - 32];
    out[i] = f2bf(v0 * cs + other * sn);
}

__global__ __launch_bounds__(256) void softmax_causal(const float* __restrict__ sc,
                                                      u16* __restrict__ pb, int S, float scale) {
    int row = blockIdx.x;          // b*NH*S + h*S + sq
    int sq  = row % S;
    const float* r = sc + (long long)row * S;
    u16* ob = pb + (long long)row * S;
    int tid = threadIdx.x;
    __shared__ float redA[8];
    __shared__ float redB[8];

    float mx = -3.0e38f;
    for (int j = tid; j <= sq; j += 256) mx = fmaxf(mx, r[j] * scale);
#pragma unroll
    for (int t = 16; t > 0; t >>= 1) mx = fmaxf(mx, __shfl_xor(mx, t));
    if ((tid & 31) == 0) redA[tid >> 5] = mx;
    __syncthreads();
    float gmx = -3.0e38f;
#pragma unroll
    for (int w = 0; w < 8; ++w) gmx = fmaxf(gmx, redA[w]);

    float s = 0.f;
    for (int j = tid; j <= sq; j += 256) s += __expf(r[j] * scale - gmx);
#pragma unroll
    for (int t = 16; t > 0; t >>= 1) s += __shfl_xor(s, t);
    if ((tid & 31) == 0) redB[tid >> 5] = s;
    __syncthreads();
    float gs = 0.f;
#pragma unroll
    for (int w = 0; w < 8; ++w) gs += redB[w];
    float inv = 1.f / gs;

    for (int j = tid; j < S; j += 256) {
        float o = (j <= sq) ? __expf(r[j] * scale - gmx) * inv : 0.f;
        ob[j] = f2bf(o);
    }
}

__global__ void add_residual(float* __restrict__ x, const float* __restrict__ t, long long n) {
    long long i = (long long)blockIdx.x * blockDim.x + threadIdx.x;
    if (i < n) x[i] += t[i];
}

__global__ void gelu_bf16(const float* __restrict__ in, u16* __restrict__ out, long long n) {
    long long i = (long long)blockIdx.x * blockDim.x + threadIdx.x;
    if (i < n) {
        float v = in[i];
        out[i] = f2bf(0.5f * v * (1.f + erff(v * 0.70710678118f)));
    }
}

// ---------------- MoE routing ----------------
__global__ void router_kernel(const u16* __restrict__ h, const float* __restrict__ Wr,
                              const float* __restrict__ br, int* __restrict__ ti,
                              float* __restrict__ tw, int N, int H) {
    int n = blockIdx.x * blockDim.x + threadIdx.x;
    if (n >= N) return;
    float acc[8];
#pragma unroll
    for (int e = 0; e < 8; ++e) acc[e] = br[e];
    const u16* hr = h + (long long)n * H;
    for (int j = 0; j < H; ++j) {
        float hv = bf2f(hr[j]);
        const float* wr = Wr + (long long)j * 8;
#pragma unroll
        for (int e = 0; e < 8; ++e) acc[e] += hv * wr[e];
    }
    float mx = acc[0];
#pragma unroll
    for (int e = 1; e < 8; ++e) mx = fmaxf(mx, acc[e]);
    float p[8], se = 0.f;
#pragma unroll
    for (int e = 0; e < 8; ++e) { p[e] = __expf(acc[e] - mx); se += p[e]; }
#pragma unroll
    for (int e = 0; e < 8; ++e) p[e] /= se;
    int i0 = 0; float p0 = p[0];
#pragma unroll
    for (int e = 1; e < 8; ++e) if (p[e] > p0) { p0 = p[e]; i0 = e; }
    int i1 = -1; float p1 = -1.f;
#pragma unroll
    for (int e = 0; e < 8; ++e) if (e != i0 && p[e] > p1) { p1 = p[e]; i1 = e; }
    float norm = 1.f / (p0 + p1 + 1e-9f);
    ti[2 * n] = i0; ti[2 * n + 1] = i1;
    tw[2 * n] = p0 * norm; tw[2 * n + 1] = p1 * norm;
}

__global__ void zero_ints(int* __restrict__ p, int n) {
    int i = blockIdx.x * blockDim.x + threadIdx.x;
    if (i < n) p[i] = 0;
}
__global__ void count_kernel(const int* __restrict__ ti, int* __restrict__ cnt, int N) {
    int n = blockIdx.x * blockDim.x + threadIdx.x;
    if (n >= N) return;
    atomicAdd(&cnt[ti[2 * n]], 1);
    atomicAdd(&cnt[ti[2 * n + 1]], 1);
}
__global__ void scan_kernel(const int* __restrict__ cnt, int* __restrict__ off,
                            int* __restrict__ wp, int E) {
    if (threadIdx.x == 0 && blockIdx.x == 0) {
        int acc = 0;
        for (int e = 0; e < E; ++e) { off[e] = acc; wp[e] = acc; acc += cnt[e]; }
    }
}
__global__ void place_kernel(const int* __restrict__ ti, const float* __restrict__ tw,
                             int* __restrict__ wp, int* __restrict__ map,
                             float* __restrict__ wv, int N) {
    int n = blockIdx.x * blockDim.x + threadIdx.x;
    if (n >= N) return;
#pragma unroll
    for (int k = 0; k < 2; ++k) {
        int e = ti[2 * n + k];
        int pos = atomicAdd(&wp[e], 1);
        map[pos] = n;
        wv[pos] = tw[2 * n + k];
    }
}
__global__ void gather_kernel(const u16* __restrict__ h, const int* __restrict__ map,
                              u16* __restrict__ aM, int H) {
    int r = blockIdx.x;
    long long src = (long long)map[r] * H, dst = (long long)r * H;
    for (int c = threadIdx.x; c < H; c += blockDim.x) aM[dst + c] = h[src + c];
}
__global__ void scatter_kernel(float* __restrict__ x, const float* __restrict__ h2,
                               const int* __restrict__ map, const float* __restrict__ wv, int H) {
    int r = blockIdx.x;
    int tok = map[r];
    float w = wv[r];
    for (int c = threadIdx.x; c < H; c += blockDim.x)
        atomicAdd(&x[(long long)tok * H + c], w * h2[(long long)r * H + c]);
}

// ---------------- host side ----------------
static inline void launch_gemm(hipStream_t st, const u16* A, const u16* B, float* C,
                               const float* bias, int M, int N, int K,
                               int lda, int ldb, int ldc, int bT, int batches, int zDiv,
                               long long sAo, long long sAi, long long sBo, long long sBi,
                               long long sCo, long long sCi, long long sBias,
                               const int* segCnt, const int* segOff) {
    dim3 g((M + 127) / 128, (N + 63) / 64, batches);
    gemm_wmma_bf16<<<g, 256, 0, st>>>(A, B, C, bias, M, N, K, lda, ldb, ldc, bT, zDiv,
                                      sAo, sAi, sBo, sBi, sCo, sCi, sBias, segCnt, segOff);
}

extern "C" void kernel_launch(void* const* d_in, const int* in_sizes, int n_in,
                              void* d_out, int out_size, void* d_ws, size_t ws_size,
                              hipStream_t stream) {
    (void)in_sizes; (void)n_in; (void)out_size; (void)ws_size;
    const int L_ = 2, B_ = 2, S_ = 1024, H_ = 768, NH_ = 12, HD_ = 64;
    const int E_ = 8, F_ = 3072, V_ = 50257, N_ = B_ * S_;
    const int NK_ = N_ * 2;                     // total routed rows (always 4096)
    const float scale = 0.125f;                 // 1/sqrt(64)

    const int*   ids   = (const int*)d_in[0];
    const float* emb   = (const float*)d_in[1];
    const float* ln1w  = (const float*)d_in[2];
    const float* ln1b  = (const float*)d_in[3];
    const float* ln2w  = (const float*)d_in[4];
    const float* ln2b  = (const float*)d_in[5];
    const float* lnfw  = (const float*)d_in[6];
    const float* lnfb  = (const float*)d_in[7];
    const float* Wq    = (const float*)d_in[8];
    const float* bq    = (const float*)d_in[9];
    const float* Wk    = (const float*)d_in[10];
    const float* bk    = (const float*)d_in[11];
    const float* Wv    = (const float*)d_in[12];
    const float* bv    = (const float*)d_in[13];
    const float* Wo    = (const float*)d_in[14];
    const float* bo    = (const float*)d_in[15];
    const float* Wr    = (const float*)d_in[16];
    const float* br    = (const float*)d_in[17];
    const float* W1    = (const float*)d_in[18];
    const float* b1    = (const float*)d_in[19];
    const float* W2    = (const float*)d_in[20];
    const float* b2    = (const float*)d_in[21];

    // ---- carve workspace ----
    char* p = (char*)d_ws;
    auto carve = [&](size_t bytes) -> void* {
        void* r = (void*)p;
        p += (bytes + 255) & ~(size_t)255;
        return r;
    };
    u16*   embB = (u16*)carve((size_t)V_ * H_ * 2);
    u16*   WqB  = (u16*)carve((size_t)L_ * H_ * H_ * 2);
    u16*   WkB  = (u16*)carve((size_t)L_ * H_ * H_ * 2);
    u16*   WvB  = (u16*)carve((size_t)L_ * H_ * H_ * 2);
    u16*   WoB  = (u16*)carve((size_t)L_ * H_ * H_ * 2);
    u16*   W1B  = (u16*)carve((size_t)L_ * E_ * H_ * F_ * 2);
    u16*   W2B  = (u16*)carve((size_t)L_ * E_ * F_ * H_ * 2);
    float* x    = (float*)carve((size_t)N_ * H_ * 4);
    u16*   hB   = (u16*)carve((size_t)N_ * H_ * 2);
    float* qF   = (float*)carve((size_t)N_ * H_ * 4);
    float* kF   = (float*)carve((size_t)N_ * H_ * 4);
    float* vF   = (float*)carve((size_t)N_ * H_ * 4);
    u16*   qB   = (u16*)carve((size_t)N_ * H_ * 2);
    u16*   kB   = (u16*)carve((size_t)N_ * H_ * 2);
    u16*   vB   = (u16*)carve((size_t)N_ * H_ * 2);
    float* sc   = (float*)carve((size_t)B_ * NH_ * S_ * S_ * 4);
    u16*   pbuf = (u16*)carve((size_t)B_ * NH_ * S_ * S_ * 2);
    float* oF   = (float*)carve((size_t)N_ * H_ * 4);
    u16*   oB   = (u16*)carve((size_t)N_ * H_ * 2);
    float* tF   = (float*)carve((size_t)N_ * H_ * 4);
    int*   ti   = (int*)carve((size_t)N_ * 2 * 4);
    float* tw   = (float*)carve((size_t)N_ * 2 * 4);
    int*   cnt  = (int*)carve(E_ * 4);
    int*   off  = (int*)carve(E_ * 4);
    int*   wp   = (int*)carve(E_ * 4);
    int*   map  = (int*)carve((size_t)NK_ * 4);
    float* wv   = (float*)carve((size_t)NK_ * 4);
    u16*   aM   = (u16*)carve((size_t)NK_ * H_ * 2);
    float* h1F  = (float*)carve((size_t)NK_ * F_ * 4);
    u16*   h1B  = (u16*)carve((size_t)NK_ * F_ * 2);
    float* h2F  = (float*)carve((size_t)NK_ * H_ * 4);
    u16*   xfB  = (u16*)carve((size_t)N_ * H_ * 2);
    float* logits = (float*)d_out;

    auto cvt = [&](const float* in, u16* out, long long n) {
        cvt_f32_bf16<<<(unsigned)((n + 255) / 256), 256, 0, stream>>>(in, out, n);
    };

    // ---- weight conversion to bf16 (each call; deterministic) ----
    cvt(emb, embB, (long long)V_ * H_);
    cvt(Wq, WqB, (long long)L_ * H_ * H_);
    cvt(Wk, WkB, (long long)L_ * H_ * H_);
    cvt(Wv, WvB, (long long)L_ * H_ * H_);
    cvt(Wo, WoB, (long long)L_ * H_ * H_);
    cvt(W1, W1B, (long long)L_ * E_ * H_ * F_);
    cvt(W2, W2B, (long long)L_ * E_ * F_ * H_);

    // ---- embedding ----
    embed_kernel<<<N_, 256, 0, stream>>>(ids, emb, x, H_);

    for (int l = 0; l < L_; ++l) {
        // === attention ===
        ln_kernel<<<N_, 32, 0, stream>>>(x, ln1w + l * H_, ln1b + l * H_, hB, H_);

        launch_gemm(stream, hB, WqB + (size_t)l * H_ * H_, qF, bq + l * H_,
                    N_, H_, H_, H_, H_, H_, 0, 1, 1, 0, 0, 0, 0, 0, 0, 0, nullptr, nullptr);
        launch_gemm(stream, hB, WkB + (size_t)l * H_ * H_, kF, bk + l * H_,
                    N_, H_, H_, H_, H_, H_, 0, 1, 1, 0, 0, 0, 0, 0, 0, 0, nullptr, nullptr);
        launch_gemm(stream, hB, WvB + (size_t)l * H_ * H_, vF, bv + l * H_,
                    N_, H_, H_, H_, H_, H_, 0, 1, 1, 0, 0, 0, 0, 0, 0, 0, nullptr, nullptr);

        long long nh = (long long)N_ * H_;
        rope_kernel<<<(unsigned)((nh + 255) / 256), 256, 0, stream>>>(qF, qB, S_, H_, nh);
        rope_kernel<<<(unsigned)((nh + 255) / 256), 256, 0, stream>>>(kF, kB, S_, H_, nh);
        cvt(vF, vB, nh);

        // scores[b,h] = Q · K^T  (batched over B*NH)
        launch_gemm(stream, qB, kB, sc, nullptr, S_, S_, HD_, H_, H_, S_, 1,
                    B_ * NH_, NH_,
                    (long long)S_ * H_, HD_, (long long)S_ * H_, HD_,
                    (long long)NH_ * S_ * S_, (long long)S_ * S_, 0, nullptr, nullptr);

        softmax_causal<<<B_ * NH_ * S_, 256, 0, stream>>>(sc, pbuf, S_, scale);

        // o[b,h] = P · V
        launch_gemm(stream, pbuf, vB, oF, nullptr, S_, HD_, S_, S_, H_, H_, 0,
                    B_ * NH_, NH_,
                    (long long)NH_ * S_ * S_, (long long)S_ * S_,
                    (long long)S_ * H_, HD_, (long long)S_ * H_, HD_, 0, nullptr, nullptr);

        cvt(oF, oB, nh);
        launch_gemm(stream, oB, WoB + (size_t)l * H_ * H_, tF, bo + l * H_,
                    N_, H_, H_, H_, H_, H_, 0, 1, 1, 0, 0, 0, 0, 0, 0, 0, nullptr, nullptr);
        add_residual<<<(unsigned)((nh + 255) / 256), 256, 0, stream>>>(x, tF, nh);

        // === MoE ===
        ln_kernel<<<N_, 32, 0, stream>>>(x, ln2w + l * H_, ln2b + l * H_, hB, H_);
        router_kernel<<<(N_ + 255) / 256, 256, 0, stream>>>(hB, Wr + (size_t)l * H_ * E_,
                                                            br + l * E_, ti, tw, N_, H_);
        zero_ints<<<1, 32, 0, stream>>>(cnt, E_);
        count_kernel<<<(N_ + 255) / 256, 256, 0, stream>>>(ti, cnt, N_);
        scan_kernel<<<1, 1, 0, stream>>>(cnt, off, wp, E_);
        place_kernel<<<(N_ + 255) / 256, 256, 0, stream>>>(ti, tw, wp, map, wv, N_);
        gather_kernel<<<NK_, 256, 0, stream>>>(hB, map, aM, H_);

        // h1 = gathered_h @ W1[e] + b1[e]   (segmented over experts)
        launch_gemm(stream, aM, W1B + (size_t)l * E_ * H_ * F_, h1F, b1 + (size_t)l * E_ * F_,
                    NK_, F_, H_, H_, F_, F_, 0, E_, 1,
                    0, 0, (long long)H_ * F_, 0, 0, 0, F_, cnt, off);
        long long n1 = (long long)NK_ * F_;
        gelu_bf16<<<(unsigned)((n1 + 255) / 256), 256, 0, stream>>>(h1F, h1B, n1);
        // h2 = gelu(h1) @ W2[e] + b2[e]
        launch_gemm(stream, h1B, W2B + (size_t)l * E_ * F_ * H_, h2F, b2 + (size_t)l * E_ * H_,
                    NK_, H_, F_, F_, H_, H_, 0, E_, 1,
                    0, 0, (long long)F_ * H_, 0, 0, 0, H_, cnt, off);
        scatter_kernel<<<NK_, 256, 0, stream>>>(x, h2F, map, wv, H_);
    }

    // === final LN + tied lm_head ===
    ln_kernel<<<N_, 32, 0, stream>>>(x, lnfw, lnfb, xfB, H_);
    launch_gemm(stream, xfB, embB, logits, nullptr, N_, V_, H_, H_, H_, V_, 1,
                1, 1, 0, 0, 0, 0, 0, 0, 0, nullptr, nullptr);
}
